// Attention_66425964200571
// MI455X (gfx1250) — matrix operand order repaired and checked by reading, not compiled, not run on hardware
//
#include <hip/hip_runtime.h>

// ---------------------------------------------------------------------------
// Transposed-channel attention block for MI455X (gfx1250), bf16 WMMA pipeline.
// D = A x B + C via v_wmma_f32_16x16x32_bf16 (wave32).
// Fragments = 2x ds_load_b128; per-iteration order: frag loads -> staging of
// next double-buffered chunk -> WMMAs (enables partial s_wait_dscnt overlap).
// ---------------------------------------------------------------------------

typedef __attribute__((ext_vector_type(16))) __bf16 bf16x16;
typedef __attribute__((ext_vector_type(8)))  float  floatx8;

union Frag { bf16x16 v; unsigned short u[16]; uint4 q[2]; };

#define BATCH 8
#define C0    192
#define HH    128
#define WW    128
#define NPIX  16384   // H*W
#define HEADS 4
#define CH    48
#define C3    576     // 3*C
#define K9    5184    // C3 * 9
#define BP    40      // transposed B-tile LDS pitch (ushorts): 80B rows, 16B-aligned

__device__ __forceinline__ unsigned short f2bf(float f) {
  unsigned int u = __float_as_uint(f);
  u += 0x7FFFu + ((u >> 16) & 1u);       // round-to-nearest-even
  return (unsigned short)(u >> 16);
}

__device__ __forceinline__ void store_out(float* p, float v) { *p = v; }
__device__ __forceinline__ void store_out(unsigned short* p, float v) { *p = f2bf(v); }

// A fragment (16x32 bf16): LDS row-major [m][32]; lane(m,hi) = run [hi*16, +16)
__device__ __forceinline__ bf16x16 load_a_frag(const unsigned short* a, int lane) {
  const int m = lane & 15, hi = (lane >> 4) & 1;
  const uint4* p = (const uint4*)(a + m * 32 + hi * 16);  // 64m + 32hi bytes: 16B aligned
  Frag f;
  f.q[0] = p[0];
  f.q[1] = p[1];                                          // +8 elements = +16B
  return f.v;
}

// B fragment (32x16 bf16) from K-transposed LDS [n][pitch]; lane(n,hi) = run [hi*16, +16)
__device__ __forceinline__ bf16x16 load_bT_frag(const unsigned short* b, int pitch, int lane) {
  const int n = lane & 15, hi = (lane >> 4) & 1;
  const uint4* p = (const uint4*)(b + n * pitch + hi * 16);
  Frag f;
  f.q[0] = p[0];
  f.q[1] = p[1];
  return f.v;
}

// ---------------------------------------------------------------------------
// fp32 -> bf16 converter, float4 -> uint2 (n must be a multiple of 4)
// ---------------------------------------------------------------------------
__global__ void cvt_bf16_kernel(const float* __restrict__ s, unsigned short* __restrict__ d,
                                size_t n4) {
  for (size_t i = (size_t)blockIdx.x * blockDim.x + threadIdx.x; i < n4;
       i += (size_t)gridDim.x * blockDim.x) {
    float4 v = ((const float4*)s)[i];
    uint2 o;
    o.x = (unsigned int)f2bf(v.x) | ((unsigned int)f2bf(v.y) << 16);
    o.y = (unsigned int)f2bf(v.z) | ((unsigned int)f2bf(v.w) << 16);
    ((uint2*)d)[i] = o;
  }
}

// ---------------------------------------------------------------------------
// Staging helpers
// ---------------------------------------------------------------------------
// GEMM A tile 16x32 from row-major A[M][K] (b32 loads/stores)
__device__ __forceinline__ void stage_gemm_A(const unsigned short* __restrict__ A,
                                             unsigned short* __restrict__ As,
                                             int m0, int K, int k0, int tid) {
  int r = tid >> 4, c2 = tid & 15;
  ((unsigned int*)As)[tid] = ((const unsigned int*)(A + (size_t)(m0 + r) * K + k0))[c2];
}

// GEMM B tile 32(K)x128(N), stored K-transposed [n][BP]; pack (k,k+1) -> b32 store
__device__ __forceinline__ void stage_gemm_B(const unsigned short* __restrict__ Bb,
                                             unsigned short* __restrict__ Bs,
                                             int N, int k0, int n0, int tid) {
#pragma unroll
  for (int s = 0; s < 8; ++s) {
    int slot = tid + 256 * s;          // 2048 slots = 128 n x 16 k-pairs
    int n  = slot & 127;
    int kp = slot >> 7;
    const unsigned short* p = Bb + (size_t)(k0 + 2 * kp) * N + n0 + n;
    unsigned int v0 = p[0];
    unsigned int v1 = p[N];
    *(unsigned int*)(Bs + n * BP + 2 * kp) = v0 | (v1 << 16);
  }
}

// Conv weight tile 128(oc)x32(k) row-major, uint4 loads/stores
__device__ __forceinline__ void stage_conv_A(const unsigned short* __restrict__ Wt,
                                             unsigned short* __restrict__ As,
                                             int oc0, int k0, int tid) {
  int r = tid >> 1, half = tid & 1;
  int oc = oc0 + r;
  uint4 v0 = uint4{0, 0, 0, 0}, v1 = uint4{0, 0, 0, 0};
  if (oc < C3) {
    const uint4* g = (const uint4*)(Wt + (size_t)oc * K9 + k0 + half * 16);
    v0 = g[0];
    v1 = g[1];
  }
  uint4* d = (uint4*)(As + r * 32 + half * 16);
  d[0] = v0;
  d[1] = v1;
}

// Conv image tile 32(k)x128(x) with 3x3 halo, stored K-transposed [x][BP]
__device__ __forceinline__ void stage_conv_B(const unsigned short* __restrict__ InB,
                                             unsigned short* __restrict__ Bs,
                                             int y, int k0, int tid) {
#pragma unroll
  for (int s = 0; s < 8; ++s) {
    int slot = tid + 256 * s;          // 2048 slots = 128 x * 16 k-pairs
    int x  = slot & 127;
    int kp = slot >> 7;
    int k  = k0 + 2 * kp;
    int ic = k / 9;                    // decompose once per pair, then increment
    int rr = k - ic * 9;
    unsigned int packed = 0;
#pragma unroll
    for (int e = 0; e < 2; ++e) {
      int d3 = rr / 3;                 // 0..2
      int dy = d3 - 1;
      int dx = rr - d3 * 3 - 1;
      int yy = y + dy;
      int xx = x + dx;
      unsigned short v = 0;
      if (yy >= 0 && yy < HH && xx >= 0 && xx < WW)
        v = InB[((size_t)ic * HH + yy) * WW + xx];
      packed |= ((unsigned int)v) << (16 * e);
      if (++rr == 9) { rr = 0; ++ic; }
    }
    *(unsigned int*)(Bs + x * BP + 2 * kp) = packed;
  }
}

// ---------------------------------------------------------------------------
// Generic GEMM: C[b] = A (MxK, shared) x B[b] (KxN).  Block = 8 waves,
// tile 16(M) x 128(N), wave owns one 16x16 N-subtile. Double-buffered LDS.
// ---------------------------------------------------------------------------
template <typename OutT>
__global__ __launch_bounds__(256) void gemm_wmma(const unsigned short* __restrict__ A,
                                                 const unsigned short* __restrict__ B,
                                                 OutT* __restrict__ C,
                                                 int M, int K, int N) {
  __shared__ unsigned short As[2][16 * 32];
  __shared__ unsigned short Bs[2][128 * BP];
  const int tid  = threadIdx.x;
  const int lane = tid & 31;
  const int wave = tid >> 5;
  const int m0   = blockIdx.x * 16;
  const int n0   = blockIdx.y * 128;
  const int b    = blockIdx.z;
  const unsigned short* Bb = B + (size_t)b * K * N;
  OutT* Cb = C + (size_t)b * M * N;

  const int nk = K / 32;
  stage_gemm_A(A, As[0], m0, K, 0, tid);
  stage_gemm_B(Bb, Bs[0], N, 0, n0, tid);

  floatx8 acc = {};
  for (int i = 0; i < nk; ++i) {
    __syncthreads();
    const int cur = i & 1;
    // 1) fragment loads from current buffer
    bf16x16 a  = load_a_frag(As[cur], lane);
    bf16x16 bf = load_bT_frag(&Bs[cur][wave * 16 * BP], BP, lane);
    // 2) stage next chunk into the other buffer
    if (i + 1 < nk) {
      stage_gemm_A(A, As[1 - cur], m0, K, (i + 1) * 32, tid);
      stage_gemm_B(Bb, Bs[1 - cur], N, (i + 1) * 32, n0, tid);
    }
    // 3) matrix op
    acc = __builtin_amdgcn_wmma_f32_16x16x32_bf16(false, a, false, bf, (short)0, acc, false, false);
  }

  const int nn  = lane & 15;
  const int hi8 = (lane < 16) ? 0 : 8;
  const int col = n0 + wave * 16 + nn;
#pragma unroll
  for (int r = 0; r < 8; ++r) {
    int row = m0 + r + hi8;
    store_out(&Cb[(size_t)row * N + col], acc[r]);
  }
}

// ---------------------------------------------------------------------------
// 3x3 conv (pad 1) as implicit GEMM. N-tile = one image row (W=128).
// Block = 8 waves -> 128(oc) x 128(x); K = 5184 in 32-chunks, double-buffered.
// B-fragments processed in two groups of 4 to bound VGPR pressure while
// overlapping group-1 ds loads with group-0 WMMAs.
// ---------------------------------------------------------------------------
__global__ __launch_bounds__(256) void conv3x3_wmma(const unsigned short* __restrict__ Wt, // [576][5184]
                                                    const unsigned short* __restrict__ In, // [B][576][128][128]
                                                    float* __restrict__ Out) {             // [B][576][128][128]
  __shared__ unsigned short As[2][128 * 32];
  __shared__ unsigned short Bs[2][128 * BP];
  const int tid  = threadIdx.x;
  const int lane = tid & 31;
  const int wave = tid >> 5;
  const int oc0  = blockIdx.x * 128;
  const int by   = blockIdx.y;
  const int b    = by >> 7;
  const int y    = by & 127;
  const unsigned short* InB = In + (size_t)b * C3 * NPIX;

  floatx8 acc[8] = {};

  const int nk = K9 / 32;   // 162
  stage_conv_A(Wt, As[0], oc0, 0, tid);
  stage_conv_B(InB, Bs[0], y, 0, tid);

  for (int i = 0; i < nk; ++i) {
    __syncthreads();
    const int cur = i & 1;
    // 1) fragment loads (group 0) from current buffer
    bf16x16 a = load_a_frag(&As[cur][wave * 16 * 32], lane);
    bf16x16 bfr0[4];
#pragma unroll
    for (int t = 0; t < 4; ++t) bfr0[t] = load_bT_frag(&Bs[cur][t * 16 * BP], BP, lane);
    // 2) stage next chunk into the other buffer (global loads + LDS stores)
    if (i + 1 < nk) {
      int pr = oc0 + (tid & 127);
      if (pr < C3 && (i + 2) < nk)
        __builtin_prefetch(&Wt[(size_t)pr * K9 + (i + 2) * 32], 0, 1);  // global_prefetch_b8
      stage_conv_A(Wt, As[1 - cur], oc0, (i + 1) * 32, tid);
      stage_conv_B(InB, Bs[1 - cur], y, (i + 1) * 32, tid);
    }
    // 3) group-0 WMMAs; group-1 loads overlap them
#pragma unroll
    for (int t = 0; t < 4; ++t)
      acc[t] = __builtin_amdgcn_wmma_f32_16x16x32_bf16(false, a, false, bfr0[t], (short)0, acc[t],
                                                       false, false);
    bf16x16 bfr1[4];
#pragma unroll
    for (int t = 0; t < 4; ++t) bfr1[t] = load_bT_frag(&Bs[cur][(4 + t) * 16 * BP], BP, lane);
#pragma unroll
    for (int t = 0; t < 4; ++t)
      acc[4 + t] = __builtin_amdgcn_wmma_f32_16x16x32_bf16(false, a, false, bfr1[t], (short)0,
                                                           acc[4 + t], false, false);
  }

  const int nn     = lane & 15;
  const int hi8    = (lane < 16) ? 0 : 8;
  const int oc_sub = oc0 + wave * 16;
  if (oc_sub < C3) {
    float* OutB = Out + (size_t)b * C3 * NPIX + (size_t)y * WW;
#pragma unroll
    for (int t = 0; t < 8; ++t) {
      int x = t * 16 + nn;
#pragma unroll
      for (int r = 0; r < 8; ++r) {
        int oc = oc_sub + r + hi8;
        OutB[(size_t)oc * NPIX + x] = acc[t][r];
      }
    }
  }
}

// ---------------------------------------------------------------------------
// Per-row L2 norm over spatial dim for q,k channels (c in [0,384) of qkv).
// ---------------------------------------------------------------------------
__global__ __launch_bounds__(256) void rownorm_kernel(const float* __restrict__ qkv,
                                                      float* __restrict__ inv) {
  const int rowid = blockIdx.x;               // b*384 + c
  const int b = rowid / 384, c = rowid % 384;
  const float4* p = (const float4*)(qkv + ((size_t)b * C3 + c) * NPIX);
  float s = 0.f;
  for (int i = threadIdx.x; i < NPIX / 4; i += 256) {
    float4 v = p[i];
    s += v.x * v.x + v.y * v.y + v.z * v.z + v.w * v.w;
  }
  __shared__ float red[256];
  red[threadIdx.x] = s;
  __syncthreads();
  for (int off = 128; off > 0; off >>= 1) {
    if (threadIdx.x < off) red[threadIdx.x] += red[threadIdx.x + off];
    __syncthreads();
  }
  if (threadIdx.x == 0) inv[rowid] = 1.0f / fmaxf(sqrtf(red[0]), 1e-12f);
}

// ---------------------------------------------------------------------------
// Normalize q,k; convert q,k,v to bf16 planes [b][192][N]. float4 -> uint2.
// ---------------------------------------------------------------------------
__global__ __launch_bounds__(256) void normconv_kernel(const float* __restrict__ qkv,
                                                       const float* __restrict__ inv,
                                                       unsigned short* __restrict__ qn,
                                                       unsigned short* __restrict__ kn,
                                                       unsigned short* __restrict__ vb) {
  const size_t total4 = (size_t)BATCH * C3 * (NPIX / 4);
  for (size_t idx = (size_t)blockIdx.x * 256 + threadIdx.x; idx < total4;
       idx += (size_t)gridDim.x * 256) {
    size_t n4 = idx % (NPIX / 4);
    size_t c  = (idx / (NPIX / 4)) % C3;
    size_t b  = idx / ((size_t)(NPIX / 4) * C3);
    float4 v = ((const float4*)qkv)[idx];
    float sc;
    unsigned short* plane;
    size_t cc;
    if (c < 192)      { sc = inv[b * 384 + c];  plane = qn; cc = c; }
    else if (c < 384) { sc = inv[b * 384 + c];  plane = kn; cc = c - 192; }
    else              { sc = 1.0f;              plane = vb; cc = c - 384; }
    uint2 o;
    o.x = (unsigned int)f2bf(v.x * sc) | ((unsigned int)f2bf(v.y * sc) << 16);
    o.y = (unsigned int)f2bf(v.z * sc) | ((unsigned int)f2bf(v.w * sc) << 16);
    ((uint2*)plane)[((size_t)b * 192 + cc) * (NPIX / 4) + n4] = o;
  }
}

// ---------------------------------------------------------------------------
// attn[b,h] = softmax(scale[h] * qn[b,h] . kn[b,h]^T)  -> bf16 [bh][48][48].
// One block per (b,h); 9 waves = 3x3 tiles; K staged double-buffered in LDS.
// ---------------------------------------------------------------------------
__device__ __forceinline__ void stage_attn(const unsigned short* __restrict__ qn,
                                           const unsigned short* __restrict__ kn,
                                           size_t base, int k0,
                                           unsigned short* __restrict__ stg, int tid) {
  for (int idx = tid; idx < 1536; idx += 288) {   // 96 rows x 16 uints
    int r = idx >> 4, c2 = idx & 15;
    const unsigned short* src =
        (r < 48) ? (qn + base + (size_t)r * NPIX) : (kn + base + (size_t)(r - 48) * NPIX);
    ((unsigned int*)stg)[idx] = ((const unsigned int*)(src + k0))[c2];
  }
}

__global__ __launch_bounds__(288) void attn_softmax_kernel(const unsigned short* __restrict__ qn,
                                                           const unsigned short* __restrict__ kn,
                                                           const float* __restrict__ scale,
                                                           unsigned short* __restrict__ attn_bf) {
  __shared__ unsigned short stg[2][96 * 32];  // rows 0-47: q, 48-95: k ([c][k])
  __shared__ float attn_s[48][49];
  const int tid  = threadIdx.x;
  const int lane = tid & 31;
  const int wave = tid >> 5;                  // 0..8
  const int bh   = blockIdx.x;
  const int b    = bh >> 2;
  const int h    = bh & 3;
  const int mi   = wave / 3;
  const int ni   = wave % 3;
  const size_t base = ((size_t)b * 192 + h * 48) * NPIX;

  const int nk = NPIX / 32;   // 512
  stage_attn(qn, kn, base, 0, stg[0], tid);

  floatx8 acc = {};
  for (int i = 0; i < nk; ++i) {
    __syncthreads();
    const int cur = i & 1;
    bf16x16 a  = load_a_frag(&stg[cur][mi * 16 * 32], lane);
    bf16x16 bf = load_bT_frag(&stg[cur][(48 + ni * 16) * 32], 32, lane);
    if (i + 1 < nk) stage_attn(qn, kn, base, (i + 1) * 32, stg[1 - cur], tid);
    acc = __builtin_amdgcn_wmma_f32_16x16x32_bf16(false, a, false, bf, (short)0, acc, false, false);
  }

  const float sc = scale[h];
  {
    const int nn  = lane & 15;
    const int hi8 = (lane < 16) ? 0 : 8;
#pragma unroll
    for (int r = 0; r < 8; ++r) attn_s[mi * 16 + r + hi8][ni * 16 + nn] = acc[r] * sc;
  }
  __syncthreads();
  if (tid < 48) {
    float mx = -3.4e38f;
    for (int j = 0; j < 48; ++j) mx = fmaxf(mx, attn_s[tid][j]);
    float sum = 0.f;
    for (int j = 0; j < 48; ++j) sum += expf(attn_s[tid][j] - mx);
    float is = 1.0f / sum;
    unsigned short* dst = attn_bf + (size_t)bh * 48 * 48 + tid * 48;
    for (int j = 0; j < 48; ++j) dst[j] = f2bf(expf(attn_s[tid][j] - mx) * is);
  }
}

// ---------------------------------------------------------------------------
// out[b,h] = attn[b,h] (48x48) x v[b,h] (48xN) -> bf16 [b][192][N].
// K = 48 zero-padded to 64 (2 WMMA chunks). 3 waves = 3 M-tiles, 8 N-subtiles.
// ---------------------------------------------------------------------------
__global__ __launch_bounds__(96) void attn_v_kernel(const unsigned short* __restrict__ attn_bf,
                                                    const unsigned short* __restrict__ vb,
                                                    unsigned short* __restrict__ oa) {
  const int tid  = threadIdx.x;
  const int lane = tid & 31;
  const int wave = tid >> 5;                 // mi = 0..2
  const int bh   = blockIdx.x;
  const int b    = bh >> 2;
  const int h    = bh & 3;
  const int n0   = blockIdx.y * 128;
  const unsigned short* Ab = attn_bf + (size_t)bh * 2304;
  const size_t vbase = ((size_t)b * 192 + h * 48) * NPIX;

  Frag a0, a1;
  {
    const int m  = lane & 15;
    const int hi = (lane >> 4) & 1;
    const unsigned short* ap = Ab + (size_t)(wave * 16 + m) * 48;  // 96B rows, 16B aligned
    a0.q[0] = *(const uint4*)(ap + hi * 16);       // K 0..7  / 16..23
    a0.q[1] = *(const uint4*)(ap + hi * 16 + 8);   // K 8..15 / 24..31
    a1.q[0] = hi ? uint4{0, 0, 0, 0} : *(const uint4*)(ap + 32);   // K 32..39 / zero pad
    a1.q[1] = hi ? uint4{0, 0, 0, 0} : *(const uint4*)(ap + 40);   // K 40..47 / zero pad
  }
#pragma unroll
  for (int nj = 0; nj < 8; ++nj) {
    const int n  = n0 + nj * 16 + (lane & 15);
    const int kb = (lane < 16) ? 0 : 16;
    Frag b0, b1;
#pragma unroll
    for (int j = 0; j < 16; ++j) b0.u[j] = vb[vbase + (size_t)(kb + j) * NPIX + n];
    if (lane < 16) {
#pragma unroll
      for (int j = 0; j < 16; ++j) b1.u[j] = vb[vbase + (size_t)(32 + j) * NPIX + n];
    } else {
      b1.q[0] = uint4{0, 0, 0, 0};
      b1.q[1] = uint4{0, 0, 0, 0};
    }
    floatx8 acc = {};
    acc = __builtin_amdgcn_wmma_f32_16x16x32_bf16(false, a0.v, false, b0.v, (short)0, acc, false, false);
    acc = __builtin_amdgcn_wmma_f32_16x16x32_bf16(false, a1.v, false, b1.v, (short)0, acc, false, false);
    const int nn  = lane & 15;
    const int hi8 = (lane < 16) ? 0 : 8;
#pragma unroll
    for (int r = 0; r < 8; ++r) {
      int c = h * 48 + wave * 16 + r + hi8;
      oa[((size_t)b * 192 + c) * NPIX + n0 + nj * 16 + nn] = f2bf(acc[r]);
    }
  }
}

// ---------------------------------------------------------------------------
extern "C" void kernel_launch(void* const* d_in, const int* in_sizes, int n_in,
                              void* d_out, int out_size, void* d_ws, size_t ws_size,
                              hipStream_t stream) {
  (void)in_sizes; (void)n_in; (void)out_size; (void)ws_size;
  const float* x     = (const float*)d_in[0];
  const float* w1    = (const float*)d_in[1];
  const float* w2    = (const float*)d_in[2];
  const float* wp    = (const float*)d_in[3];
  const float* scale = (const float*)d_in[4];

  char* ws = (char*)d_ws;
  size_t off = 0;
  auto alloc = [&](size_t bytes) -> char* {
    char* p = ws + off;
    off += (bytes + 255) & ~(size_t)255;
    return p;
  };

  unsigned short* x_bf    = (unsigned short*)alloc((size_t)BATCH * C0 * NPIX * 2); // reused as oa
  unsigned short* qkv1_bf = (unsigned short*)alloc((size_t)BATCH * C3 * NPIX * 2); // reused: qn/kn/v
  float*          qkv_f   = (float*)alloc((size_t)BATCH * C3 * NPIX * 4);
  unsigned short* w1_bf   = (unsigned short*)alloc((size_t)C3 * C0 * 2);
  unsigned short* w2_bf   = (unsigned short*)alloc((size_t)C3 * K9 * 2);
  unsigned short* wp_bf   = (unsigned short*)alloc((size_t)C0 * C0 * 2);
  float*          inv_nrm = (float*)alloc((size_t)BATCH * 384 * 4);
  unsigned short* attn_bf = (unsigned short*)alloc((size_t)32 * 48 * 48 * 2);

  // aliases (source buffers dead by the time these are written)
  unsigned short* qn = qkv1_bf;
  unsigned short* kn = qkv1_bf + (size_t)BATCH * C0 * NPIX;
  unsigned short* vb = qkv1_bf + (size_t)BATCH * 2 * C0 * NPIX;
  unsigned short* oa = x_bf;

  // 1) converts (element counts all divisible by 4)
  cvt_bf16_kernel<<<2048, 256, 0, stream>>>(x,  x_bf,  (size_t)BATCH * C0 * NPIX / 4);
  cvt_bf16_kernel<<<108,  256, 0, stream>>>(w1, w1_bf, (size_t)C3 * C0 / 4);
  cvt_bf16_kernel<<<1024, 256, 0, stream>>>(w2, w2_bf, (size_t)C3 * K9 / 4);
  cvt_bf16_kernel<<<36,   256, 0, stream>>>(wp, wp_bf, (size_t)C0 * C0 / 4);

  // 2) qkv1 = w1 (576x192) @ x (192xN) per batch -> bf16
  gemm_wmma<unsigned short><<<dim3(C3 / 16, NPIX / 128, BATCH), 256, 0, stream>>>(
      w1_bf, x_bf, qkv1_bf, C3, C0, NPIX);

  // 3) qkv = conv3x3(qkv1) -> f32 (dominant: ~783 GFLOP)
  conv3x3_wmma<<<dim3((C3 + 127) / 128, BATCH * HH), 256, 0, stream>>>(w2_bf, qkv1_bf, qkv_f);

  // 4) norms + normalize/convert
  rownorm_kernel<<<BATCH * 384, 256, 0, stream>>>(qkv_f, inv_nrm);
  normconv_kernel<<<4096, 256, 0, stream>>>(qkv_f, inv_nrm, qn, kn, vb);

  // 5) attn = softmax(scale * qn kn^T)
  attn_softmax_kernel<<<32, 288, 0, stream>>>(qn, kn, scale, attn_bf);

  // 6) oa = attn @ v
  attn_v_kernel<<<dim3(32, NPIX / 128), 96, 0, stream>>>(attn_bf, vb, oa);

  // 7) out = wp (192x192) @ oa per batch -> f32
  gemm_wmma<float><<<dim3(C0 / 16, NPIX / 128, BATCH), 256, 0, stream>>>(
      wp_bf, oa, (float*)d_out, C0, C0, NPIX);
}
